// align_union_16020228014676
// MI455X (gfx1250) — compile-verified, matrix operand order checked by reading, not run on hardware
//
#include <hip/hip_runtime.h>
#include <hip/hip_bf16.h>

// ---------------------------------------------------------------------------
// 2-layer relational GAT for MI455X (gfx1250, wave32).
//   - dense GEMMs via v_wmma_f32_16x16x32_f16 (f16 in, f32 accumulate)
//   - A fragments built directly from global with aligned b128 loads
//   - B (weights) pre-packed once per launch to transposed/padded f16
//   - edge attention / segment sums via wave32 butterflies + f32 atomics
//     (entity table = 102 MB -> resident in the 192 MB global L2)
// ---------------------------------------------------------------------------

#define ED   128          // E_DIM
#define KGR  1000         // relations
#define BETA 0.3f
#define LALPHA 0.2f

typedef _Float16 v16h __attribute__((ext_vector_type(16)));
typedef _Float16 v8h  __attribute__((ext_vector_type(8)));
typedef float    v8f  __attribute__((ext_vector_type(8)));
typedef float    v4f  __attribute__((ext_vector_type(4)));

// f32x16 -> f16x16 fragment (4 float4 runs -> one WMMA A operand)
__device__ __forceinline__ v16h cvt_frag(v4f a, v4f b, v4f c, v4f d)
{
    v16h r;
    r[0]  = (_Float16)a[0]; r[1]  = (_Float16)a[1];
    r[2]  = (_Float16)a[2]; r[3]  = (_Float16)a[3];
    r[4]  = (_Float16)b[0]; r[5]  = (_Float16)b[1];
    r[6]  = (_Float16)b[2]; r[7]  = (_Float16)b[3];
    r[8]  = (_Float16)c[0]; r[9]  = (_Float16)c[1];
    r[10] = (_Float16)c[2]; r[11] = (_Float16)c[3];
    r[12] = (_Float16)d[0]; r[13] = (_Float16)d[1];
    r[14] = (_Float16)d[2]; r[15] = (_Float16)d[3];
    return r;
}

// 8 N-tiles of WMMA against pre-packed transposed weights Bt[n][Kpad]
__device__ __forceinline__ void mma8(v8f acc[8], v16h afrag,
                                     const _Float16* __restrict__ Bt,
                                     int Kpad, int l16, int kofs)
{
    #pragma unroll
    for (int nt = 0; nt < 8; ++nt) {
        const _Float16* bp = Bt + (size_t)(nt * 16 + l16) * Kpad + kofs;
        v8h b0 = *(const v8h*)bp;          // K run {0..7} (or {8..15})
        v8h b1 = *(const v8h*)(bp + 16);   // K run {16..23} (or {24..31})
        v16h bfrag = __builtin_shufflevector(b0, b1,
            0,1,2,3,4,5,6,7,8,9,10,11,12,13,14,15);
        acc[nt] = __builtin_amdgcn_wmma_f32_16x16x32_f16(
            false, afrag, false, bfrag, (short)0, acc[nt], false, false);
    }
}

// ---------------------------------------------------------------------------
// WMMA GEMM:  C[M x 128] = A[M x K] @ B[K x 128] (+ bias[n] if bias != null)
// Block = 256 threads (8 waves); tile = 128 rows x 128 cols; K-step = 32.
// Bt is B transposed to [128 x Kpad] f16, zero-padded to Kpad (mult of 32).
// ---------------------------------------------------------------------------
__global__ __launch_bounds__(256)
void wmma_gemm128(const float* __restrict__ A, const _Float16* __restrict__ Bt,
                  const float* __restrict__ bias, float* __restrict__ C,
                  int M, int K, int Kpad)
{
    const int tid  = threadIdx.x;
    const int wave = tid >> 5;
    const int lane = tid & 31;
    const int half = lane >> 4;        // lane 0-15 vs 16-31
    const int l16  = lane & 15;
    const int kb0  = half * 8;         // K runs {0..7,16..23} / {8..15,24..31}

    const int mRow = blockIdx.x * 128 + wave * 16 + l16;
    const int mld  = mRow < M ? mRow : M - 1;       // clamp: loads stay valid
    const float* __restrict__ arow = A + (size_t)mld * K;

    v8f acc[8] = {};

    const int kfull = K & ~31;
    int k0 = 0;
    for (; k0 < kfull; k0 += 32) {
        if (k0 + 32 < kfull) __builtin_prefetch(arow + k0 + 32, 0, 1);
        const int ka = k0 + kb0;
        v4f a0 = *(const v4f*)(arow + ka);
        v4f a1 = *(const v4f*)(arow + ka + 4);
        v4f a2 = *(const v4f*)(arow + ka + 16);
        v4f a3 = *(const v4f*)(arow + ka + 20);
        mma8(acc, cvt_frag(a0, a1, a2, a3), Bt, Kpad, l16, ka);
    }
    if (k0 < K) {   // tail step (K % 32 != 0); Bt is zero-padded there
        const int ka = k0 + kb0;
        const v4f z = {0.f, 0.f, 0.f, 0.f};
        v4f a0 = (ka      + 4 <= K) ? *(const v4f*)(arow + ka)      : z;
        v4f a1 = (ka + 4  + 4 <= K) ? *(const v4f*)(arow + ka + 4)  : z;
        v4f a2 = (ka + 16 + 4 <= K) ? *(const v4f*)(arow + ka + 16) : z;
        v4f a3 = (ka + 20 + 4 <= K) ? *(const v4f*)(arow + ka + 20) : z;
        mma8(acc, cvt_frag(a0, a1, a2, a3), Bt, Kpad, l16, ka);
    }

    // store C (+bias); 16x16 f32 C layout: M = r + 8*half, N = l16.
    // a wave's 32 stores form two contiguous 64B row segments.
    #pragma unroll
    for (int nt = 0; nt < 8; ++nt) {
        const int n = nt * 16 + l16;
        const float bv = bias ? bias[n] : 0.0f;
        #pragma unroll
        for (int r = 0; r < 8; ++r) {
            const int gm = blockIdx.x * 128 + wave * 16 + r + half * 8;
            if (gm < M) C[(size_t)gm * ED + n] = acc[nt][r] + bv;
        }
    }
}

// ---------------------------------------------------------------------------
// pack weights: B[K x 128] f32 -> Bt[128 x Kpad] f16 (transposed, 0-padded)
// ---------------------------------------------------------------------------
__global__ void pack_w(const float* __restrict__ B, _Float16* __restrict__ Bt,
                       int K, int Kpad)
{
    int idx = blockIdx.x * blockDim.x + threadIdx.x;
    if (idx >= ED * Kpad) return;
    int n = idx / Kpad;
    int k = idx - n * Kpad;
    Bt[idx] = (_Float16)(k < K ? B[(size_t)k * ED + n] : 0.0f);
}

// ---------------------------------------------------------------------------
// zero fill
// ---------------------------------------------------------------------------
__global__ void zero_f32(float* __restrict__ p, size_t n)
{
    size_t i = (size_t)blockIdx.x * blockDim.x + threadIdx.x;
    if (i < n) p[i] = 0.0f;
}

// ---------------------------------------------------------------------------
// relation segment sums: Lr[r] += Le[h], Rr[r] += Re[t], cnt[r] += 1
// one thread per (triple, dim); dim is the fast (coalesced) index
// ---------------------------------------------------------------------------
__global__ __launch_bounds__(256)
void r_segment(const float* __restrict__ Le, const float* __restrict__ Re,
               const int* __restrict__ th, const int* __restrict__ tr,
               const int* __restrict__ tt, int nT,
               float* __restrict__ Lr, float* __restrict__ Rr,
               float* __restrict__ cnt)
{
    size_t idx = (size_t)blockIdx.x * blockDim.x + threadIdx.x;
    if (idx >= (size_t)nT * ED) return;
    int i = (int)(idx >> 7);
    int d = (int)(idx & (ED - 1));
    int r = tr[i];
    atomicAdd(&Lr[(size_t)r * ED + d], Le[(size_t)th[i] * ED + d]);
    atomicAdd(&Rr[(size_t)r * ED + d], Re[(size_t)tt[i] * ED + d]);
    if (d == 0) atomicAdd(&cnt[r], 1.0f);
}

// r_layer = relu([Lr,Rr] * safe_recip(cnt))
__global__ void r_finalize(const float* __restrict__ Lr, const float* __restrict__ Rr,
                           const float* __restrict__ cnt, float* __restrict__ r_layer)
{
    int idx = blockIdx.x * blockDim.x + threadIdx.x;
    if (idx >= KGR * ED) return;
    int r = idx >> 7, d = idx & (ED - 1);
    float c = cnt[r];
    float inv = (c == 0.0f) ? 0.0f : 1.0f / c;
    float a = Lr[idx] * inv;
    float b = Rr[idx] * inv;
    r_layer[(size_t)r * (2 * ED) + d]      = a > 0.0f ? a : 0.0f;
    r_layer[(size_t)r * (2 * ED) + ED + d] = b > 0.0f ? b : 0.0f;
}

// ---------------------------------------------------------------------------
// edge attention: one wave32 per directed edge.
// att = exp(-leaky( dot(concat(ent[row],ent[col]) * r_layer[rel], w_att) ))
// rowsum[row] += att ; e_acc[row,:] += att * ent[col,:]
// ---------------------------------------------------------------------------
__global__ __launch_bounds__(256)
void edge_att(const float* __restrict__ ent, const float* __restrict__ r_layer,
              const float* __restrict__ w_att,
              const int* __restrict__ th, const int* __restrict__ tr,
              const int* __restrict__ tt, int nT,
              float* __restrict__ rowsum, float* __restrict__ e_acc)
{
    const int e    = blockIdx.x * 8 + (threadIdx.x >> 5);
    const int lane = threadIdx.x & 31;
    const int nE   = 2 * nT;
    if (e >= nE) return;

    int row, col, rel;
    if (e < nT) { row = th[e];      col = tt[e];      rel = tr[e]; }
    else        { row = tt[e - nT]; col = th[e - nT]; rel = tr[e - nT]; }

    const float* er = ent + (size_t)row * ED;
    const float* ec = ent + (size_t)col * ED;
    const float* rl = r_layer + (size_t)rel * (2 * ED);

    float s = 0.0f;
    #pragma unroll
    for (int j = 0; j < 4; ++j) {
        int d = lane + 32 * j;
        s += er[d] * rl[d]      * w_att[d];
        s += ec[d] * rl[ED + d] * w_att[ED + d];
    }
    // wave32 butterfly reduction: every lane ends with the full sum
    #pragma unroll
    for (int m = 16; m >= 1; m >>= 1) s += __shfl_xor(s, m, 32);

    float l   = (s >= 0.0f) ? s : LALPHA * s;
    float att = expf(-l);

    if (lane == 0) atomicAdd(&rowsum[row], att);
    #pragma unroll
    for (int j = 0; j < 4; ++j) {
        int d = lane + 32 * j;
        atomicAdd(&e_acc[(size_t)row * ED + d], att * ec[d]);
    }
}

// out = base + beta * relu(e_acc * safe_recip(rowsum))
__global__ void e_finalize(const float* __restrict__ base, const float* __restrict__ e_acc,
                           const float* __restrict__ rowsum, float beta,
                           float* __restrict__ out, int E)
{
    size_t idx = (size_t)blockIdx.x * blockDim.x + threadIdx.x;
    if (idx >= (size_t)E * ED) return;
    int i = (int)(idx >> 7);
    float rs  = rowsum[i];
    float inv = (rs == 0.0f) ? 0.0f : 1.0f / rs;
    float v   = e_acc[idx] * inv;
    v = v > 0.0f ? v : 0.0f;
    out[idx] = base[idx] + beta * v;
}

// ---------------------------------------------------------------------------
// launch
// ---------------------------------------------------------------------------
extern "C" void kernel_launch(void* const* d_in, const int* in_sizes, int n_in,
                              void* d_out, int out_size, void* d_ws, size_t ws_size,
                              hipStream_t stream)
{
    const float* kg_name_embed = (const float*)d_in[0];  // [E,300]
    const float* kg_name_w     = (const float*)d_in[1];  // [300,128]
    const float* kg_name_b     = (const float*)d_in[2];  // [128,1]
    const float* w_L           = (const float*)d_in[3];  // [128,128]
    const float* w_R           = (const float*)d_in[4];  // [128,128]
    const float* w_att         = (const float*)d_in[5];  // [256,1]
    const int*   th            = (const int*)d_in[6];
    const int*   tr            = (const int*)d_in[7];
    const int*   tt            = (const int*)d_in[8];

    const int E  = in_sizes[0] / 300;
    const int nT = in_sizes[6];
    const size_t Ed = (size_t)E * ED;

    const int Kname = 300, KnamePad = 320;

    // workspace layout (f32 region)
    float* ws         = (float*)d_ws;
    float* name_embed = ws;                       // E*128
    float* gat1       = name_embed + Ed;          // E*128
    float* Le         = gat1 + Ed;                // E*128
    float* Re         = Le + Ed;                  // E*128
    float* r_layer    = Re + Ed;                  // R*256
    float* Lr         = r_layer + (size_t)KGR * 2 * ED;  // R*128  --+
    float* Rr         = Lr + (size_t)KGR * ED;           // R*128    | contiguous
    float* cnt        = Rr + (size_t)KGR * ED;           // R        | zero region
    float* rowsum     = cnt + KGR;                       // E        |
    float* e_acc      = rowsum + E;                      // E*128  --+
    const size_t nzero = (size_t)KGR * ED * 2 + KGR + E + Ed;

    // f16 packed-weights region (16B aligned; sizes are multiples of 8 halves)
    size_t fused = (size_t)(e_acc - ws) + Ed;
    fused = (fused + 7) & ~(size_t)7;
    _Float16* Bt_name = (_Float16*)(ws + fused);            // 128*320
    _Float16* BtL     = Bt_name + (size_t)ED * KnamePad;    // 128*128
    _Float16* BtR     = BtL + (size_t)ED * ED;              // 128*128

    float* gat_out = (float*)d_out;

    const dim3 blk(256);
    const dim3 gGemm((E + 127) / 128);
    const dim3 gZero((unsigned)((nzero + 255) / 256));
    const dim3 gSeg((unsigned)(((size_t)nT * ED + 255) / 256));
    const dim3 gRFin((KGR * ED + 255) / 256);
    const dim3 gEdge((2 * nT + 7) / 8);
    const dim3 gEFin((unsigned)((Ed + 255) / 256));
    const dim3 gPackN((ED * KnamePad + 255) / 256);
    const dim3 gPackE((ED * ED + 255) / 256);

    // pack weights once (tiny; Bt buffers stay hot in L2 for all GEMM blocks)
    pack_w<<<gPackN, blk, 0, stream>>>(kg_name_w, Bt_name, Kname, KnamePad);
    pack_w<<<gPackE, blk, 0, stream>>>(w_L, BtL, ED, ED);
    pack_w<<<gPackE, blk, 0, stream>>>(w_R, BtR, ED, ED);

    // name_embed = A @ W + b  (f16 WMMA, f32 accumulate)
    wmma_gemm128<<<gGemm, blk, 0, stream>>>(kg_name_embed, Bt_name, kg_name_b,
                                            name_embed, E, Kname, KnamePad);

    // ---------------- layer 1 ----------------
    zero_f32<<<gZero, blk, 0, stream>>>(Lr, nzero);
    wmma_gemm128<<<gGemm, blk, 0, stream>>>(name_embed, BtL, nullptr, Le, E, ED, ED);
    wmma_gemm128<<<gGemm, blk, 0, stream>>>(name_embed, BtR, nullptr, Re, E, ED, ED);
    r_segment<<<gSeg, blk, 0, stream>>>(Le, Re, th, tr, tt, nT, Lr, Rr, cnt);
    r_finalize<<<gRFin, blk, 0, stream>>>(Lr, Rr, cnt, r_layer);
    edge_att<<<gEdge, blk, 0, stream>>>(name_embed, r_layer, w_att,
                                        th, tr, tt, nT, rowsum, e_acc);
    e_finalize<<<gEFin, blk, 0, stream>>>(name_embed, e_acc, rowsum, BETA, gat1, E);

    // ---------------- layer 2 ----------------
    zero_f32<<<gZero, blk, 0, stream>>>(Lr, nzero);
    wmma_gemm128<<<gGemm, blk, 0, stream>>>(gat1, BtL, nullptr, Le, E, ED, ED);
    wmma_gemm128<<<gGemm, blk, 0, stream>>>(gat1, BtR, nullptr, Re, E, ED, ED);
    r_segment<<<gSeg, blk, 0, stream>>>(Le, Re, th, tr, tt, nT, Lr, Rr, cnt);
    r_finalize<<<gRFin, blk, 0, stream>>>(Lr, Rr, cnt, r_layer);
    edge_att<<<gEdge, blk, 0, stream>>>(gat1, r_layer, w_att,
                                        th, tr, tt, nT, rowsum, e_acc);
    e_finalize<<<gEFin, blk, 0, stream>>>(name_embed, e_acc, rowsum, BETA, gat_out, E);
}
